// ImprovedAttention_23072564314636
// MI455X (gfx1250) — matrix operand order repaired and checked by reading, not compiled, not run on hardware
//
#include <hip/hip_runtime.h>
#include <stdint.h>

typedef __bf16 bf16_t;
typedef __attribute__((ext_vector_type(16))) __bf16 v16bf;
typedef __attribute__((ext_vector_type(8)))  float  v8f;

union FragU {
  v16bf v;
  uint32_t u[8];
};

// A-matrix fragment (16x32 bf16, row-major src: element (m,k) at base[m*lda+k]).
// Lanes 0-15 hold M=lane, K=0..15; lanes 16-31 hold K=16..31.
static __device__ inline v16bf load_frag_A(const bf16_t* __restrict__ base, int lda, int lane) {
  const int lr = lane & 15, lg = lane >> 4;
  FragU f;
  const bf16_t* row = base + (size_t)lr * lda;
  #pragma unroll
  for (int j = 0; j < 8; ++j) {
    const int koff = 2 * j + 16 * lg;
    f.u[j] = *reinterpret_cast<const uint32_t*>(row + koff);
  }
  return f.v;
}

// B-matrix fragment (32x16 bf16), stored n-major/k-contiguous: element (k,n) at base[n*ldn+k].
// Lane = column n; lanes 0-15 hold K=0..15, lanes 16-31 hold K=16..31 (2 per VGPR).
static __device__ inline v16bf load_frag_B(const bf16_t* __restrict__ base, int ldn, int lane) {
  const int lr = lane & 15, lg = lane >> 4;
  FragU f;
  const bf16_t* col = base + (size_t)lr * ldn + 16 * lg;
  #pragma unroll
  for (int j = 0; j < 8; ++j)
    f.u[j] = *reinterpret_cast<const uint32_t*>(col + 2 * j);
  return f.v;
}

// 16-byte async global->LDS copy (CDNA5 async path, tracked by ASYNCcnt).
static __device__ inline void async_copy16(uint32_t lds_off, const void* gptr) {
  asm volatile("global_load_async_to_lds_b128 %0, %1, off"
               :: "v"(lds_off), "v"((uint64_t)(uintptr_t)gptr)
               : "memory");
}

// ---------------------------------------------------------------------------
// Kernel 1: QKV projection  C[4096x3072] = X[4096x1024] @ W[1024x3072]
// fp32 in, bf16 WMMA with f32 accumulate, fp32 out to workspace.
// ---------------------------------------------------------------------------
#define BM 128
#define BN 128
#define BKK 32
#define KP 36   // padded LDS K-stride (even -> u32-aligned pairs, spreads banks)

__global__ __launch_bounds__(256)
void qkv_gemm_kernel(const float* __restrict__ X, const float* __restrict__ W,
                     float* __restrict__ C) {
  __shared__ __align__(16) bf16_t As[BM * KP];
  __shared__ __align__(16) bf16_t Bs[BN * KP];

  const int tid  = threadIdx.x;
  const int lane = tid & 31;
  const int wave = tid >> 5;
  const int wm   = wave >> 1;   // 0..3 : 32-row strip
  const int wn   = wave & 1;    // 0..1 : 64-col strip
  const int m0   = blockIdx.y * BM;
  const int n0   = blockIdx.x * BN;
  const int lr   = lane & 15, lg = lane >> 4;

  v8f acc[2][4] = {};

  for (int kb = 0; kb < 1024; kb += BKK) {
    // Stage A tile (128x32), fp32 -> bf16, row-major.
    {
      const int row = tid >> 1;
      const int c0  = (tid & 1) * 16;
      const float* src = X + (size_t)(m0 + row) * 1024 + kb + c0;
      bf16_t* dst = As + row * KP + c0;
      #pragma unroll
      for (int i = 0; i < 16; i += 4) {
        float4 v = *reinterpret_cast<const float4*>(src + i);
        dst[i + 0] = (bf16_t)v.x; dst[i + 1] = (bf16_t)v.y;
        dst[i + 2] = (bf16_t)v.z; dst[i + 3] = (bf16_t)v.w;
      }
      if (kb + BKK < 1024) __builtin_prefetch(src + BKK, 0, 3);  // keep near (WGP)
    }
    // Stage B tile (32k x 128n), transposed into n-major/k-contiguous LDS.
    {
      const int k  = tid >> 3;
      const int nb = (tid & 7) * 16;
      const float* src = W + (size_t)(kb + k) * 3072 + n0 + nb;
      #pragma unroll
      for (int i = 0; i < 16; i += 4) {
        float4 v = *reinterpret_cast<const float4*>(src + i);
        Bs[(nb + i + 0) * KP + k] = (bf16_t)v.x;
        Bs[(nb + i + 1) * KP + k] = (bf16_t)v.y;
        Bs[(nb + i + 2) * KP + k] = (bf16_t)v.z;
        Bs[(nb + i + 3) * KP + k] = (bf16_t)v.w;
      }
      if (kb + BKK < 1024) __builtin_prefetch(src + (size_t)BKK * 3072, 0, 3);
    }
    __syncthreads();

    v16bf a0 = load_frag_A(As + (wm * 32 + 0)  * KP, KP, lane);
    v16bf a1 = load_frag_A(As + (wm * 32 + 16) * KP, KP, lane);
    #pragma unroll
    for (int nt = 0; nt < 4; ++nt) {
      v16bf b = load_frag_B(Bs + (wn * 64 + nt * 16) * KP, KP, lane);
      acc[0][nt] = __builtin_amdgcn_wmma_f32_16x16x32_bf16(false, a0, false, b,
                     (short)0, acc[0][nt], false, false);
      acc[1][nt] = __builtin_amdgcn_wmma_f32_16x16x32_bf16(false, a1, false, b,
                     (short)0, acc[1][nt], false, false);
    }
    __syncthreads();
  }

  // Epilogue: fp32 store (C/D layout: VGPR r -> row r+8*lg, col = lane%16).
  #pragma unroll
  for (int mt = 0; mt < 2; ++mt)
    #pragma unroll
    for (int nt = 0; nt < 4; ++nt)
      #pragma unroll
      for (int r = 0; r < 8; ++r) {
        const size_t row = (size_t)(m0 + wm * 32 + mt * 16 + r + 8 * lg);
        const int    col = n0 + wn * 64 + nt * 16 + lr;
        C[row * 3072 + col] = acc[mt][nt][r];
      }
}

// ---------------------------------------------------------------------------
// Kernel 2: RoPE (full-dim inv_freq variant) + pack to bf16.
// Q,K -> [B,H,S,D]; V -> transposed [B,H,D,S] (so PV B-frags are k-contiguous).
// ---------------------------------------------------------------------------
__global__ __launch_bounds__(256)
void rope_pack_kernel(const float* __restrict__ qkv, bf16_t* __restrict__ Qb,
                      bf16_t* __restrict__ Kb, bf16_t* __restrict__ Vt) {
  const int idx = blockIdx.x * blockDim.x + threadIdx.x;  // B*S*H*D = 4194304
  const int d = idx & 31;
  const int h = (idx >> 5) & 31;
  const int s = (idx >> 10) & 2047;
  const int b = idx >> 21;

  const size_t row  = (size_t)(b * 2048 + s) * 3072;
  const int    col  = h * 32 + d;
  const int    colp = h * 32 + (d ^ 16);   // rotate partner

  const float q  = qkv[row + col],         qp = qkv[row + colp];
  const float k  = qkv[row + 1024 + col],  kp = qkv[row + 1024 + colp];
  const float v  = qkv[row + 2048 + col];

  // inv_freq = 10000^(-d/32); ln(10000)/32 = 0.28782313662425574
  const float ang = (float)s * __expf(-(float)d * 0.28782313662425574f);
  float sn, cs;
  __sincosf(ang, &sn, &cs);
  const float sgn = (d < 16) ? -1.0f : 1.0f;

  const float qo = q * cs + sgn * qp * sn;
  const float ko = k * cs + sgn * kp * sn;

  const int bh = b * 32 + h;
  Qb[((size_t)bh * 2048 + s) * 32 + d] = (bf16_t)qo;
  Kb[((size_t)bh * 2048 + s) * 32 + d] = (bf16_t)ko;
  Vt[((size_t)bh * 32 + d) * 2048 + s] = (bf16_t)v;
}

// ---------------------------------------------------------------------------
// Kernel 3: causal flash attention, 8 waves/WG (128 q rows).
// K/V staged cooperatively into LDS with async global->LDS copies, double
// buffered (ASYNCcnt). Stage trip count is workgroup-uniform (qc+1 stages of
// 128 keys); causal cut is a wave-uniform early-exit over 32-key sub-blocks.
// Intra-wave LDS ordering for softmax uses s_wait_dscnt.
// ---------------------------------------------------------------------------
#define KTP 40    // K-tile row stride (elements): 80B rows, 16B aligned, bank stride 20
#define VTP 136   // V-tile row stride (elements): 272B rows, 16B aligned, bank stride 4/lane

__global__ __launch_bounds__(256)
void attn_kernel(const bf16_t* __restrict__ Q, const bf16_t* __restrict__ K,
                 const bf16_t* __restrict__ Vt, float* __restrict__ O) {
  __shared__ __align__(16) bf16_t Ktile[2][128 * KTP];  // [key][d], padded
  __shared__ __align__(16) bf16_t Vtile[2][32 * VTP];   // [d][key], padded
  __shared__ __align__(16) float  sc[8][16 * 33];       // per-wave scores
  __shared__ __align__(16) bf16_t pb[8][16 * 32];       // per-wave bf16 P tile

  const int tid  = threadIdx.x;
  const int lane = tid & 31;
  const int w    = tid >> 5;
  const int lr   = lane & 15, lg = lane >> 4;

  const int bh  = blockIdx.x >> 4;   // b*32+h
  const int qc  = blockIdx.x & 15;   // q chunk of 128
  const int b   = bh >> 5, h = bh & 31;
  const int qs0 = qc * 128 + w * 16;

  const bf16_t* Qp = Q  + ((size_t)bh * 2048 + qs0) * 32;
  const bf16_t* Kp = K  + (size_t)bh * 2048 * 32;
  const bf16_t* Vp = Vt + (size_t)bh * 32 * 2048;

  // Issue one 128-key stage: K block 128x32 (8KB) + V block 32x128 (8KB).
  // 4 async b128 instructions per wave per stage.
  auto issue_stage = [&](int st, int buf) {
    const int k0 = st * 128;
    #pragma unroll
    for (int c = 0; c < 2; ++c) {                 // K: 512 x 16B chunks
      const int chunk = tid + 256 * c;
      const int row = chunk >> 2, part = chunk & 3;
      const uint32_t lds = (uint32_t)(uintptr_t)&Ktile[buf][row * KTP] + part * 16;
      async_copy16(lds, Kp + (size_t)(k0 + row) * 32 + part * 8);
    }
    #pragma unroll
    for (int c = 0; c < 2; ++c) {                 // V: 512 x 16B chunks
      const int chunk = tid + 256 * c;
      const int row = chunk >> 4, part = chunk & 15;
      const uint32_t lds = (uint32_t)(uintptr_t)&Vtile[buf][row * VTP] + part * 16;
      async_copy16(lds, Vp + (size_t)row * 2048 + k0 + part * 8);
    }
  };

  const v16bf aq = load_frag_A(Qp, 32, lane);   // Q frag lives in regs all loop

  v8f o0 = {}, o1 = {};
  float m_run = -INFINITY, l_run = 0.0f;
  const float scale = 0.17677669529663687f;     // 1/sqrt(32)

  float*  scw = sc[w];
  bf16_t* pbw = pb[w];

  const int nst = qc + 1;                       // workgroup-uniform stage count
  issue_stage(0, 0);

  for (int st = 0; st < nst; ++st) {
    const int buf = st & 1;
    if (st + 1 < nst) {
      issue_stage(st + 1, buf ^ 1);             // overlap next stage with compute
      asm volatile("s_wait_asynccnt 0x4" ::: "memory");  // stage st complete
    } else {
      asm volatile("s_wait_asynccnt 0x0" ::: "memory");
    }
    __syncthreads();

    for (int sub = 0; sub < 4; ++sub) {
      const int k0g = st * 128 + sub * 32;
      if (k0g > qs0 + 15) break;                // wave-uniform causal cut

      // ---- scores: two 16x16 WMMA tiles from LDS K ----
      #pragma unroll
      for (int t = 0; t < 2; ++t) {
        v16bf bk = load_frag_B(&Ktile[buf][(sub * 32 + t * 16) * KTP], KTP, lane);
        v8f s = {};
        s = __builtin_amdgcn_wmma_f32_16x16x32_bf16(false, aq, false, bk,
              (short)0, s, false, false);
        const int key = k0g + t * 16 + lr;
        #pragma unroll
        for (int r = 0; r < 8; ++r) {
          const int sq = qs0 + r + 8 * lg;
          float v = s[r] * scale;
          if (key > sq) v = -INFINITY;          // causal mask
          scw[(r + 8 * lg) * 33 + t * 16 + lr] = v;
        }
      }
      asm volatile("s_wait_dscnt 0" ::: "memory");

      // ---- online softmax: lane handles row (lane&15), half (lane>>4) ----
      const int rbase = lr * 33 + lg * 16;
      float mloc = -INFINITY;
      #pragma unroll
      for (int c = 0; c < 16; ++c) mloc = fmaxf(mloc, scw[rbase + c]);
      mloc = fmaxf(mloc, __shfl_xor(mloc, 16, 32));
      const float m_new = fmaxf(m_run, mloc);
      const float fac   = __expf(m_run - m_new);
      float psum = 0.0f;
      #pragma unroll
      for (int c = 0; c < 16; ++c) {
        const float p = __expf(scw[rbase + c] - m_new);
        pbw[lr * 32 + lg * 16 + c] = (bf16_t)p;
        psum += p;
      }
      psum += __shfl_xor(psum, 16, 32);
      l_run = l_run * fac + psum;
      m_run = m_new;
      asm volatile("s_wait_dscnt 0" ::: "memory");

      // ---- rescale output accumulators (row factor broadcast via shfl) ----
      #pragma unroll
      for (int r = 0; r < 8; ++r) {
        const float f = __shfl(fac, r + 8 * lg, 32);
        o0[r] *= f;
        o1[r] *= f;
      }

      // ---- PV: P(16x32) @ V(32x32) as two 16-col WMMA tiles from LDS ----
      v16bf ap  = load_frag_A(pbw, 32, lane);
      v16bf bv0 = load_frag_B(&Vtile[buf][sub * 32], VTP, lane);            // d 0..15
      v16bf bv1 = load_frag_B(&Vtile[buf][16 * VTP + sub * 32], VTP, lane); // d 16..31
      o0 = __builtin_amdgcn_wmma_f32_16x16x32_bf16(false, ap, false, bv0,
             (short)0, o0, false, false);
      o1 = __builtin_amdgcn_wmma_f32_16x16x32_bf16(false, ap, false, bv1,
             (short)0, o1, false, false);
    }
    __syncthreads();   // all waves done with buf before it is re-filled
  }

  // ---- normalize and store fp32 output [B,S,H*D] ----
  #pragma unroll
  for (int r = 0; r < 8; ++r) {
    const float li = 1.0f / __shfl(l_run, r + 8 * lg, 32);
    const int   sq = qs0 + r + 8 * lg;
    float* orow = O + ((size_t)b * 2048 + sq) * 1024 + h * 32;
    orow[lr]      = o0[r] * li;
    orow[16 + lr] = o1[r] * li;
  }
}

// ---------------------------------------------------------------------------
extern "C" void kernel_launch(void* const* d_in, const int* in_sizes, int n_in,
                              void* d_out, int out_size, void* d_ws, size_t ws_size,
                              hipStream_t stream) {
  const float* x = (const float*)d_in[0];   // [2,2048,1024]
  const float* W = (const float*)d_in[1];   // [1024,3072]
  // d_in[2]: causal mask -- computed analytically, not read.
  float* out = (float*)d_out;               // [2,2048,1024]

  char* ws = (char*)d_ws;
  float*  qkv = (float*)ws;                                   // 4096*3072 f32
  bf16_t* Qb  = (bf16_t*)(ws + (size_t)4096 * 3072 * 4);      // [B,H,S,D] bf16
  bf16_t* Kb  = Qb + (size_t)4096 * 1024;                     // [B,H,S,D] bf16
  bf16_t* Vt  = Kb + (size_t)4096 * 1024;                     // [B,H,D,S] bf16

  qkv_gemm_kernel<<<dim3(3072 / BN, 4096 / BM), 256, 0, stream>>>(x, W, qkv);
  rope_pack_kernel<<<(2 * 2048 * 32 * 32) / 256, 256, 0, stream>>>(qkv, Qb, Kb, Vt);
  attn_kernel<<<2 * 32 * 16, 256, 0, stream>>>(Qb, Kb, Vt, out);
}
